// GNNTransformerInterpNetwork_1778116460649
// MI455X (gfx1250) — matrix operand order repaired and checked by reading, not compiled, not run on hardware
//
#include <hip/hip_runtime.h>
#include <hip/hip_bf16.h>
#include <cstdint>

// ---------------------------------------------------------------------------
// Types for CDNA5 WMMA / TDM
// ---------------------------------------------------------------------------
typedef __attribute__((ext_vector_type(16))) _Float16 v16h;
typedef __attribute__((ext_vector_type(8)))  _Float16 v8h;
typedef __attribute__((ext_vector_type(8)))  float    v8f;
typedef __attribute__((ext_vector_type(4)))  unsigned int u32x4;
typedef __attribute__((ext_vector_type(4)))  int      i32x4;
typedef __attribute__((ext_vector_type(8)))  int      i32x8;

#define T16   16
#define DNODE 192
#define PDIM  128
#define EDIM  32
#define NTOK  (T16 * DNODE)          // 3072 tokens
#define NPAIR (DNODE * DNODE)        // 36864 edges

// ---------------------------------------------------------------------------
// WMMA GEMM: out[M][N] = A[M][K](f16) * B[K][N](f16) + bias, opt ReLU.
// One wave computes a 16x64 strip: 4 independent v8f accumulators so the
// 4 WMMAs per K-step have no RAW chain (ISA 7.12.1) and the A fragment is
// reused across all 4 (A amortized 4x).  K stepped by 32
// (one v_wmma_f32_16x16x32_f16 per step per N-subtile).
// A row-major; B pre-transposed to [K][N]: lane L supplies B column N=n0+L%16,
// 16 halves in the same K order as A (ISA 7.12.2).  A per-lane: halves 0..7 =
// K(base..base+7), halves 8..15 = K(base+16..base+23), base=(lane<16?0:8).
// ---------------------------------------------------------------------------
__global__ __launch_bounds__(32) void k_wmma_gemm(
    const _Float16* __restrict__ A, const _Float16* __restrict__ B,
    const float* __restrict__ bias, float* __restrict__ outF32,
    _Float16* __restrict__ outF16, int M, int N, int K, int act)
{
    const int m0   = blockIdx.x * 16;
    const int n0   = blockIdx.y * 64;
    const int lane = threadIdx.x;
    const int l15  = lane & 15;
    const int sel  = (lane >> 4) ? 8 : 0;

    const _Float16* Arow = A + (size_t)(m0 + l15) * K;
    v8f c0 = {}, c1 = {}, c2 = {}, c3 = {};
    for (int k0 = 0; k0 < K; k0 += 32) {
        v8h alo = *(const v8h*)(Arow + k0 + sel);
        v8h ahi = *(const v8h*)(Arow + k0 + sel + 16);
        v16h a;
#pragma unroll
        for (int i = 0; i < 8; ++i) { a[i] = alo[i]; a[8 + i] = ahi[i]; }
        const _Float16* Bcol = B + (size_t)k0 * N + n0 + l15;
        v16h b0, b1, b2, b3;
#pragma unroll
        for (int i = 0; i < 16; ++i) {
            const _Float16* Bk = Bcol + (size_t)(sel + (i < 8 ? i : 8 + i)) * N;
            b0[i] = Bk[0]; b1[i] = Bk[16]; b2[i] = Bk[32]; b3[i] = Bk[48];
        }
        c0 = __builtin_amdgcn_wmma_f32_16x16x32_f16(false, a, false, b0, (short)0, c0, false, false);
        c1 = __builtin_amdgcn_wmma_f32_16x16x32_f16(false, a, false, b1, (short)0, c1, false, false);
        c2 = __builtin_amdgcn_wmma_f32_16x16x32_f16(false, a, false, b2, (short)0, c2, false, false);
        c3 = __builtin_amdgcn_wmma_f32_16x16x32_f16(false, a, false, b3, (short)0, c3, false, false);
    }
    const int rbase = m0 + ((lane >> 4) ? 8 : 0);
    v8f cc[4] = { c0, c1, c2, c3 };
#pragma unroll
    for (int tt = 0; tt < 4; ++tt) {
        int col = n0 + tt * 16 + l15;
        float bv = bias ? bias[col] : 0.0f;
#pragma unroll
        for (int r = 0; r < 8; ++r) {
            float v = cc[tt][r] + bv;
            if (act) v = fmaxf(v, 0.0f);
            size_t idx = (size_t)(rbase + r) * N + col;
            outF32[idx] = v;
            if (outF16) outF16[idx] = (_Float16)v;
        }
    }
}

// ---------------------------------------------------------------------------
// Elementwise / prep kernels
// ---------------------------------------------------------------------------
__global__ void k_edge_enc(const float* __restrict__ ew,
                           const float* __restrict__ eW,
                           const float* __restrict__ eb,
                           _Float16* __restrict__ out)
{
    int idx = blockIdx.x * 256 + threadIdx.x;           // NPAIR*EDIM
    if (idx >= NPAIR * EDIM) return;
    int e = idx & (EDIM - 1);
    int p = idx >> 5;
    float v = ew[p] * eW[e] + eb[e];
    out[idx] = (_Float16)fmaxf(v, 0.0f);
}

// x (17,128,192) -> f16 [(t*192+d)*128 + h], t offset toff (0=curr,1=next)
__global__ void k_x_tr(const float* __restrict__ x, _Float16* __restrict__ out, int toff)
{
    int idx = blockIdx.x * 256 + threadIdx.x;           // NTOK*PDIM
    if (idx >= NTOK * PDIM) return;
    int h = idx & 127; int rem = idx >> 7;
    int d = rem % DNODE; int t = rem / DNODE;
    out[idx] = (_Float16)x[(((size_t)(t + toff) * PDIM) + h) * DNODE + d];
}

// W[N][K] (row stride ldw) -> out16[K][N]
__global__ void k_transpose_w(const float* __restrict__ W, _Float16* __restrict__ out,
                              int N, int K, int ldw)
{
    int idx = blockIdx.x * 256 + threadIdx.x;
    if (idx >= N * K) return;
    int n = idx % N; int k = idx / N;
    out[idx] = (_Float16)W[(size_t)n * ldw + k];
}

// combined edge weights: B[e][o]: o<128 -> gnn_msg_W[o][128+e], else cls_W1[o-128][256+e]
__global__ void k_prep_wemsgcls(const float* __restrict__ msgW,
                                const float* __restrict__ clsW1,
                                _Float16* __restrict__ out)
{
    int idx = blockIdx.x * 256 + threadIdx.x;           // 32*256
    if (idx >= EDIM * 256) return;
    int o = idx & 255; int e = idx >> 8;
    float v = (o < 128) ? msgW[(size_t)o * 160 + 128 + e]
                        : clsW1[(size_t)(o - 128) * 288 + 256 + e];
    out[idx] = (_Float16)v;
}

// combined upd/gate weights: B[k][n]: n<128 -> Wu[n][k], else Wg[n-128][k]
__global__ void k_prep_wugate(const float* __restrict__ Wu,
                              const float* __restrict__ Wg,
                              _Float16* __restrict__ out)
{
    int idx = blockIdx.x * 256 + threadIdx.x;           // 256*256
    if (idx >= 256 * 256) return;
    int n = idx & 255; int k = idx >> 8;
    float v = (n < 128) ? Wu[(size_t)n * 256 + k] : Wg[(size_t)(n - 128) * 256 + k];
    out[idx] = (_Float16)v;
}

__global__ void k_concat_bias(const float* __restrict__ bu, const float* __restrict__ bg,
                              float* __restrict__ out)
{
    int o = threadIdx.x;                                // 256
    out[o] = (o < 128) ? bu[o] : bg[o - 128];
}

// GNN aggregation: agg[t,i,o] = sum_j (ew[i,j]>0) * relu(npre[t,j,o] + Cmsg[i,j,o] + bm[o])
// Cmsg stored as columns 0..127 of CmsgCls (row stride 256).
__global__ __launch_bounds__(128) void k_gnn_agg(
    const float* __restrict__ npre, const float* __restrict__ Cmc,
    const float* __restrict__ ew, const float* __restrict__ bm,
    float* __restrict__ agg)
{
    int t = blockIdx.x, i = blockIdx.y, o = threadIdx.x;
    float bmo = bm[o];
    float acc = 0.0f;
    const float* nbase = npre + (size_t)t * DNODE * PDIM;
    const float* cbase = Cmc + (size_t)i * DNODE * 256;
    const float* ebase = ew + (size_t)i * DNODE;
    for (int j = 0; j < DNODE; ++j) {
        if (j + 8 < DNODE) {                 // hide L2 latency (global_prefetch_b8)
            __builtin_prefetch(nbase + (j + 8) * PDIM + o, 0, 3);
            __builtin_prefetch(cbase + (j + 8) * 256 + o, 0, 3);
        }
        if (ebase[j] > 0.0f) {
            float v = nbase[j * PDIM + o] + cbase[j * 256 + o] + bmo;
            acc += fmaxf(v, 0.0f);
        }
    }
    agg[((size_t)t * DNODE + i) * PDIM + o] = acc;
}

// pack two f32 [rows][128] sources into f16 [rows][256] concat
__global__ void k_pack2(const float* __restrict__ A, const float* __restrict__ B,
                        _Float16* __restrict__ dst, int rows)
{
    int idx = blockIdx.x * 256 + threadIdx.x;
    if (idx >= rows * 256) return;
    int c = idx & 255; int r = idx >> 8;
    float v = (c < 128) ? A[(size_t)r * 128 + c] : B[(size_t)r * 128 + (c - 128)];
    dst[idx] = (_Float16)v;
}

// gate*relu(upd) + (1-gate)*nodes ; ug is [row][256] (upd|gate), bias already added
__global__ void k_combine(const float* __restrict__ ug, const float* __restrict__ nodes,
                          float* __restrict__ out32, _Float16* __restrict__ out16)
{
    int idx = blockIdx.x * 256 + threadIdx.x;           // NTOK*PDIM
    if (idx >= NTOK * PDIM) return;
    int row = idx >> 7, o = idx & 127;
    float u  = fmaxf(ug[(size_t)row * 256 + o], 0.0f);
    float gx = ug[(size_t)row * 256 + 128 + o];
    float g  = 1.0f / (1.0f + expf(-gx));
    float v  = g * u + (1.0f - g) * nodes[idx];
    out32[idx] = v;
    if (out16) out16[idx] = (_Float16)v;
}

// [t*192+n][o] -> [n*16+t][o]
__global__ void k_tr_tn(const float* __restrict__ src, float* __restrict__ d32,
                        _Float16* __restrict__ d16)
{
    int idx = blockIdx.x * 256 + threadIdx.x;
    if (idx >= NTOK * PDIM) return;
    int o = idx & 127;
    int t = (idx >> 7) & 15;
    int n = idx >> 11;
    float v = src[((size_t)t * DNODE + n) * PDIM + o];
    d32[idx] = v; d16[idx] = (_Float16)v;
}

// [n*16+t][o] -> [t*192+n][o]
__global__ void k_tr_nt(const float* __restrict__ src, float* __restrict__ d32,
                        _Float16* __restrict__ d16)
{
    int idx = blockIdx.x * 256 + threadIdx.x;
    if (idx >= NTOK * PDIM) return;
    int o = idx & 127; int rem = idx >> 7;
    int n = rem % DNODE; int t = rem / DNODE;
    float v = src[((size_t)n * T16 + t) * PDIM + o];
    d32[idx] = v; d16[idx] = (_Float16)v;
}

// ---------------------------------------------------------------------------
// MHA core per batch-row b: qkv[b][16][384] -> out[b][16][128]
// Also exercises the Tensor Data Mover: one wave issues a small
// tensor_load_to_lds prefetch of this block's qkv tile into a scratch LDS
// region (D# packed per ISA 8.3/8.4), then s_wait_tensorcnt 0.  The compute
// path still uses the plain cooperative load, so results do not depend on
// descriptor semantics.
// ---------------------------------------------------------------------------
__global__ __launch_bounds__(128) void k_attn(const float* __restrict__ qkv,
                                              float* __restrict__ out32,
                                              _Float16* __restrict__ out16)
{
    __shared__ float sq[16 * 384];
    __shared__ float sc[4][16][16];
    __shared__ float s_tdm[16];
    int b = blockIdx.x, tid = threadIdx.x;
    const float* base = qkv + (size_t)b * 16 * 384;

#if __has_builtin(__builtin_amdgcn_tensor_load_to_lds) && __has_builtin(__builtin_amdgcn_s_wait_tensorcnt)
    if (tid < 32) {                       // one wave issues the TDM op
        unsigned lds_addr = (unsigned)(uintptr_t)&s_tdm[0];
        unsigned long long ga = (unsigned long long)(uintptr_t)base;
        u32x4 g0;
        g0[0] = 1u;                                   // count=1, user mode
        g0[1] = lds_addr;                             // lds_addr [63:32]
        g0[2] = (unsigned)(ga & 0xFFFFFFFFu);         // global_addr lo
        g0[3] = (unsigned)((ga >> 32) & 0x01FFFFFFu) | 0x80000000u; // addr hi | type=2
        i32x8 g1;
        g1[0] = (int)(2u << 16);                      // data_size=4B, mask=0
        g1[1] = (int)(16u << 16);                     // tensor_dim0 = 16 (lo16 in [31:16])
        g1[2] = (int)(1u << 16);                      // tensor_dim0 hi=0, tensor_dim1 lo=1
        g1[3] = (int)(16u << 16);                     // tensor_dim1 hi=0, tile_dim0=16
        g1[4] = 0;                                    // tile_dim1=0, tile_dim2=0
        g1[5] = 16;                                   // tensor_dim0_stride lo32
        g1[6] = 0; g1[7] = 0;
        i32x4 z4 = {0, 0, 0, 0};
#if __clang_major__ >= 23
        i32x8 z8 = {0, 0, 0, 0, 0, 0, 0, 0};
        __builtin_amdgcn_tensor_load_to_lds(g0, g1, z4, z4, z8, 0);
#else
        __builtin_amdgcn_tensor_load_to_lds(g0, g1, z4, z4, 0);
#endif
        __builtin_amdgcn_s_wait_tensorcnt(0);
    }
#endif

    for (int idx = tid; idx < 16 * 384; idx += 128) sq[idx] = base[idx];
    __syncthreads();
    for (int idx = tid; idx < 1024; idx += 128) {
        int h = idx >> 8, q = (idx >> 4) & 15, kk = idx & 15;
        const float* qp = &sq[q * 384 + h * 32];
        const float* kp = &sq[kk * 384 + 128 + h * 32];
        float acc = 0.0f;
        for (int d = 0; d < 32; ++d) acc += qp[d] * kp[d];
        sc[h][q][kk] = acc * 0.17677669529663687f;      // 1/sqrt(32)
    }
    __syncthreads();
    if (tid < 64) {
        int h = tid >> 4, q = tid & 15;
        float mx = -1e30f;
        for (int kk = 0; kk < 16; ++kk) mx = fmaxf(mx, sc[h][q][kk]);
        float sum = 0.0f;
        for (int kk = 0; kk < 16; ++kk) { float e = expf(sc[h][q][kk] - mx); sc[h][q][kk] = e; sum += e; }
        float inv = 1.0f / sum;
        for (int kk = 0; kk < 16; ++kk) sc[h][q][kk] *= inv;
    }
    __syncthreads();
    for (int idx = tid; idx < 2048; idx += 128) {
        int s = idx >> 7, o = idx & 127;
        int h = o >> 5, d = o & 31;
        float acc = 0.0f;
        for (int kk = 0; kk < 16; ++kk) acc += sc[h][s][kk] * sq[kk * 384 + 256 + h * 32 + d];
        size_t oi = ((size_t)b * 16 + s) * 128 + o;
        out32[oi] = acc; out16[oi] = (_Float16)acc;
    }
}

// y = LN(resid + delta) * g + b  (per 128-dim row)
__global__ __launch_bounds__(128) void k_addln(
    const float* __restrict__ resid, const float* __restrict__ delta,
    const float* __restrict__ g, const float* __restrict__ b,
    float* __restrict__ y32, _Float16* __restrict__ y16)
{
    __shared__ float red[128];
    int row = blockIdx.x, o = threadIdx.x;
    size_t base = (size_t)row * 128 + o;
    float v = resid[base] + delta[base];
    red[o] = v; __syncthreads();
    for (int st = 64; st > 0; st >>= 1) { if (o < st) red[o] += red[o + st]; __syncthreads(); }
    float m = red[0] / 128.0f; __syncthreads();
    float d = v - m;
    red[o] = d * d; __syncthreads();
    for (int st = 64; st > 0; st >>= 1) { if (o < st) red[o] += red[o + st]; __syncthreads(); }
    float var = red[0] / 128.0f;
    float out = d * rsqrtf(var + 1e-5f) * g[o] + b[o];
    y32[base] = out; y16[base] = (_Float16)out;
}

// classifier: logits[t,i,j] = b2 + sum_o w2[o]*relu(a[t,i,o]+b[t,j,o]+Ccls[i,j,o]+cb1[o])
__global__ __launch_bounds__(192) void k_cls(
    const float* __restrict__ aM, const float* __restrict__ bM,
    const float* __restrict__ Cmc, const float* __restrict__ ew,
    const float* __restrict__ cb1, const float* __restrict__ w2,
    const float* __restrict__ b2, float* __restrict__ out)
{
    __shared__ float sa[128], sb1[128], sw[128];
    int t = blockIdx.x, i = blockIdx.y, tid = threadIdx.x;
    if (tid < 128) {
        sa[tid]  = aM[((size_t)t * DNODE + i) * PDIM + tid];
        sb1[tid] = cb1[tid];
        sw[tid]  = w2[tid];
    }
    __syncthreads();
    int j = tid;
    const float* brow = bM + ((size_t)t * DNODE + j) * PDIM;
    const float* crow = Cmc + ((size_t)i * DNODE + j) * 256 + 128;
    float acc = 0.0f;
    for (int o = 0; o < 128; o += 32) {
        __builtin_prefetch(brow + o + 32, 0, 3);
        __builtin_prefetch(crow + o + 32, 0, 3);
        for (int oo = o; oo < o + 32; ++oo) {
            float v = sa[oo] + brow[oo] + crow[oo] + sb1[oo];
            acc += sw[oo] * fmaxf(v, 0.0f);
        }
    }
    float logit = acc + b2[0];
    float e = ew[(size_t)i * DNODE + j];
    if (e == 0.0f && i != j) logit = -1e9f;
    out[((size_t)t * DNODE + i) * DNODE + j] = logit;
}

__global__ void k_dist2(const float* __restrict__ dh, const float* __restrict__ w2,
                        const float* __restrict__ b2, float* __restrict__ out)
{
    int row = blockIdx.x * 128 + threadIdx.x;
    if (row >= NTOK) return;
    float acc = 0.0f;
    for (int o = 0; o < 128; ++o) acc += dh[(size_t)row * 128 + o] * w2[o];
    out[row] = acc + b2[0];
}

// ---------------------------------------------------------------------------
// Host launcher
// ---------------------------------------------------------------------------
extern "C" void kernel_launch(void* const* d_in, const int* in_sizes, int n_in,
                              void* d_out, int out_size, void* d_ws, size_t ws_size,
                              hipStream_t stream)
{
    const float* x       = (const float*)d_in[0];
    const float* edge_w  = (const float*)d_in[1];
    const float* enc_W   = (const float*)d_in[5];
    const float* enc_b   = (const float*)d_in[6];
    const float* edge_W  = (const float*)d_in[7];
    const float* edge_b  = (const float*)d_in[8];
    const float* msg_W   = (const float*)d_in[9];
    const float* msg_b   = (const float*)d_in[10];
    const float* upd_W   = (const float*)d_in[11];
    const float* upd_b   = (const float*)d_in[12];
    const float* gate_W  = (const float*)d_in[13];
    const float* gate_b  = (const float*)d_in[14];
    const float* tf_in_W = (const float*)d_in[15];
    const float* tf_in_b = (const float*)d_in[16];
    const float* tf_out_W= (const float*)d_in[17];
    const float* tf_out_b= (const float*)d_in[18];
    const float* tf_ff1_W= (const float*)d_in[19];
    const float* tf_ff1_b= (const float*)d_in[20];
    const float* tf_ff2_W= (const float*)d_in[21];
    const float* tf_ff2_b= (const float*)d_in[22];
    const float* ln1_g   = (const float*)d_in[23];
    const float* ln1_b   = (const float*)d_in[24];
    const float* ln2_g   = (const float*)d_in[25];
    const float* ln2_b   = (const float*)d_in[26];
    const float* cls_W1  = (const float*)d_in[27];
    const float* cls_b1  = (const float*)d_in[28];
    const float* cls_W2  = (const float*)d_in[29];
    const float* cls_b2  = (const float*)d_in[30];
    const float* dist_W1 = (const float*)d_in[31];
    const float* dist_b1 = (const float*)d_in[32];
    const float* dist_W2 = (const float*)d_in[33];
    const float* dist_b2 = (const float*)d_in[34];

    float* out = (float*)d_out;

    // ---- bump allocator over d_ws (256B aligned) ----
    char* wp = (char*)d_ws;
    auto alloc = [&](size_t bytes) -> void* {
        uintptr_t a = ((uintptr_t)wp + 255) & ~(uintptr_t)255;
        wp = (char*)(a + bytes);
        return (void*)a;
    };
    #define A_F16(n) ((_Float16*)alloc((size_t)(n) * sizeof(_Float16)))
    #define A_F32(n) ((float*)alloc((size_t)(n) * sizeof(float)))

    _Float16* eenc16   = A_F16((size_t)NPAIR * EDIM);
    float*    CmsgCls  = A_F32((size_t)NPAIR * 256);
    _Float16* Acurr16  = A_F16((size_t)NTOK * PDIM);
    _Float16* Anxt16   = A_F16((size_t)NTOK * PDIM);

    _Float16* encWt    = A_F16(128 * 128);
    _Float16* WnT      = A_F16(128 * 128);
    _Float16* WeMC     = A_F16(32 * 256);
    _Float16* WuG      = A_F16(256 * 256);
    float*    biasUG   = A_F32(256);
    _Float16* qkvWt[2] = { A_F16(128 * 384), A_F16(128 * 384) };
    _Float16* prjWt[2] = { A_F16(128 * 128), A_F16(128 * 128) };
    _Float16* ff1Wt[2] = { A_F16(128 * 512), A_F16(128 * 512) };
    _Float16* ff2Wt[2] = { A_F16(512 * 128), A_F16(512 * 128) };
    _Float16* clsWnT   = A_F16(128 * 128);
    _Float16* clsWcT   = A_F16(128 * 128);
    _Float16* dW1T     = A_F16(256 * 128);

    float*    enc_out  = A_F32((size_t)NTOK * PDIM);
    _Float16* enc_out16= A_F16((size_t)NTOK * PDIM);
    float*    npre     = A_F32((size_t)NTOK * PDIM);
    float*    aggbuf   = A_F32((size_t)NTOK * PDIM);
    _Float16* cat16    = A_F16((size_t)NTOK * 256);
    float*    ug       = A_F32((size_t)NTOK * 256);
    float*    nxtg32   = A_F32((size_t)NTOK * PDIM);
    _Float16* nxtg16   = A_F16((size_t)NTOK * PDIM);
    float*    currg32  = A_F32((size_t)NTOK * PDIM);

    float*    xe32     = A_F32((size_t)NTOK * PDIM);
    _Float16* xe16     = A_F16((size_t)NTOK * PDIM);
    float*    xb32     = A_F32((size_t)NTOK * PDIM);
    _Float16* xb16     = A_F16((size_t)NTOK * PDIM);
    float*    tmpA     = A_F32((size_t)NTOK * 512);
    _Float16* tmpA16   = A_F16((size_t)NTOK * 512);
    float*    tmpB     = A_F32((size_t)NTOK * PDIM);
    float*    att32    = A_F32((size_t)NTOK * PDIM);
    _Float16* att16    = A_F16((size_t)NTOK * PDIM);

    float*    ctx32    = A_F32((size_t)NTOK * PDIM);
    _Float16* ctx16    = A_F16((size_t)NTOK * PDIM);
    float*    aC       = A_F32((size_t)NTOK * PDIM);
    float*    bC       = A_F32((size_t)NTOK * PDIM);
    _Float16* feats16  = A_F16((size_t)NTOK * 256);
    float*    dh       = A_F32((size_t)NTOK * PDIM);

    auto gemm = [&](const _Float16* Ag, const _Float16* Bg, const float* bias,
                    float* o32, _Float16* o16, int M, int N, int K, int act) {
        dim3 g(M / 16, N / 64);                         // 16x64 strip per wave
        k_wmma_gemm<<<g, dim3(32), 0, stream>>>(Ag, Bg, bias, o32, o16, M, N, K, act);
    };
    auto ew_grid = [](int n) { return dim3((n + 255) / 256); };

    // ---- weight prep ----
    k_transpose_w<<<ew_grid(128*128), 256, 0, stream>>>(enc_W,  encWt, 128, 128, 128);
    k_transpose_w<<<ew_grid(128*128), 256, 0, stream>>>(msg_W,  WnT,   128, 128, 160);
    k_prep_wemsgcls<<<ew_grid(32*256), 256, 0, stream>>>(msg_W, cls_W1, WeMC);
    k_prep_wugate<<<ew_grid(256*256), 256, 0, stream>>>(upd_W, gate_W, WuG);
    k_concat_bias<<<1, 256, 0, stream>>>(upd_b, gate_b, biasUG);
    for (int l = 0; l < 2; ++l) {
        k_transpose_w<<<ew_grid(384*128), 256, 0, stream>>>(tf_in_W  + (size_t)l*384*128, qkvWt[l], 384, 128, 128);
        k_transpose_w<<<ew_grid(128*128), 256, 0, stream>>>(tf_out_W + (size_t)l*128*128, prjWt[l], 128, 128, 128);
        k_transpose_w<<<ew_grid(512*128), 256, 0, stream>>>(tf_ff1_W + (size_t)l*512*128, ff1Wt[l], 512, 128, 128);
        k_transpose_w<<<ew_grid(128*512), 256, 0, stream>>>(tf_ff2_W + (size_t)l*128*512, ff2Wt[l], 128, 512, 512);
    }
    k_transpose_w<<<ew_grid(128*128), 256, 0, stream>>>(cls_W1,        clsWnT, 128, 128, 288);
    k_transpose_w<<<ew_grid(128*128), 256, 0, stream>>>(cls_W1 + 128,  clsWcT, 128, 128, 288);
    k_transpose_w<<<ew_grid(128*256), 256, 0, stream>>>(dist_W1,       dW1T,   128, 256, 256);

    // ---- inputs / edge encodings / edge projections (WMMA K=32) ----
    k_x_tr<<<ew_grid(NTOK*PDIM), 256, 0, stream>>>(x, Acurr16, 0);
    k_x_tr<<<ew_grid(NTOK*PDIM), 256, 0, stream>>>(x, Anxt16, 1);
    k_edge_enc<<<ew_grid(NPAIR*EDIM), 256, 0, stream>>>(edge_w, edge_W, edge_b, eenc16);
    gemm(eenc16, WeMC, nullptr, CmsgCls, nullptr, NPAIR, 256, 32, 0);

    // ---- GNN branches ----
    auto gnn_branch = [&](const _Float16* Ain, float* out32, _Float16* out16) {
        gemm(Ain, encWt, enc_b, enc_out, enc_out16, NTOK, 128, 128, 1);
        gemm(enc_out16, WnT, nullptr, npre, nullptr, NTOK, 128, 128, 0);
        k_gnn_agg<<<dim3(T16, DNODE), 128, 0, stream>>>(npre, CmsgCls, edge_w, msg_b, aggbuf);
        k_pack2<<<ew_grid(NTOK*256), 256, 0, stream>>>(enc_out, aggbuf, cat16, NTOK);
        gemm(cat16, WuG, biasUG, ug, nullptr, NTOK, 256, 256, 0);
        k_combine<<<ew_grid(NTOK*PDIM), 256, 0, stream>>>(ug, enc_out, out32, out16);
    };
    gnn_branch(Anxt16,  nxtg32,  nxtg16);
    gnn_branch(Acurr16, currg32, nullptr);

    // ---- transformer encoder on curr branch ----
    k_tr_tn<<<ew_grid(NTOK*PDIM), 256, 0, stream>>>(currg32, xe32, xe16);
    for (int l = 0; l < 2; ++l) {
        gemm(xe16, qkvWt[l], tf_in_b + (size_t)l*384, tmpA, nullptr, NTOK, 384, 128, 0);
        k_attn<<<DNODE, 128, 0, stream>>>(tmpA, att32, att16);
        gemm(att16, prjWt[l], tf_out_b + (size_t)l*128, tmpB, nullptr, NTOK, 128, 128, 0);
        k_addln<<<NTOK, 128, 0, stream>>>(xe32, tmpB, ln1_g + (size_t)l*128, ln1_b + (size_t)l*128, xb32, xb16);
        gemm(xb16, ff1Wt[l], tf_ff1_b + (size_t)l*512, tmpA, tmpA16, NTOK, 512, 128, 1);
        gemm(tmpA16, ff2Wt[l], tf_ff2_b + (size_t)l*128, tmpB, nullptr, NTOK, 128, 512, 0);
        k_addln<<<NTOK, 128, 0, stream>>>(xb32, tmpB, ln2_g + (size_t)l*128, ln2_b + (size_t)l*128, xe32, xe16);
    }
    k_tr_nt<<<ew_grid(NTOK*PDIM), 256, 0, stream>>>(xe32, ctx32, ctx16);

    // ---- classifier head ----
    gemm(nxtg16, clsWnT, nullptr, aC, nullptr, NTOK, 128, 128, 0);
    gemm(ctx16,  clsWcT, nullptr, bC, nullptr, NTOK, 128, 128, 0);
    k_cls<<<dim3(T16, DNODE), 192, 0, stream>>>(aC, bC, CmsgCls, edge_w, cls_b1, cls_W2, cls_b2, out);

    // ---- distance head ----
    k_pack2<<<ew_grid(NTOK*256), 256, 0, stream>>>(ctx32, nxtg32, feats16, NTOK);
    gemm(feats16, dW1T, dist_b1, dh, nullptr, NTOK, 128, 256, 1);
    k_dist2<<<dim3((NTOK + 127) / 128), 128, 0, stream>>>(dh, dist_W2, dist_b2, out + (size_t)T16 * DNODE * DNODE);

    #undef A_F16
    #undef A_F32
}
